// QuantExaone4Attention_46162308498025
// MI455X (gfx1250) — 8-bit matrix operand order repaired and checked by reading, compiled, not run on hardware
//
#include <hip/hip_runtime.h>
#include <stdint.h>

#define SQ  2048
#define HIDN 4096
#define NH  32
#define NKV 8
#define DH  128

typedef __attribute__((ext_vector_type(8))) int v8i;

__device__ __forceinline__ v8i v8i_zero() {
  v8i z = {0, 0, 0, 0, 0, 0, 0, 0};
  return z;
}

__device__ __forceinline__ float qscale(float amax) {
  return fmaxf(amax, 1e-5f) * (1.0f / 127.0f);
}

__device__ __forceinline__ v8i load_a_frag_i8(const int8_t* rowp, int h) {
  v8i a;
#pragma unroll
  for (int v = 0; v < 8; ++v) {
    int kb = ((v >> 1) << 4) + ((v & 1) << 2) + (h << 3);
    a[v] = *(const int*)(rowp + kb);
  }
  return a;
}
__device__ __forceinline__ v8i load_b_frag_i8(const int8_t* colp, int h) {
  v8i b;
#pragma unroll
  for (int v = 0; v < 8; ++v) {
    int kb = ((v >> 1) << 4) + ((v & 1) << 2) + (h << 3);
    b[v] = *(const int*)(colp + kb);
  }
  return b;
}

__global__ void init_scalars_kernel(unsigned* sc) {
  if (threadIdx.x < 16) sc[threadIdx.x] = 0u;
}

__global__ void absmax_kernel(const float* __restrict__ x, long long n,
                              unsigned* __restrict__ out) {
  float m = 0.f;
  long long stride = (long long)gridDim.x * blockDim.x;
  for (long long i = (long long)blockIdx.x * blockDim.x + threadIdx.x; i < n; i += stride)
    m = fmaxf(m, fabsf(x[i]));
#pragma unroll
  for (int off = 16; off; off >>= 1) m = fmaxf(m, __shfl_xor(m, off, 32));
  if ((threadIdx.x & 31) == 0) atomicMax(out, __float_as_uint(m));
}

__global__ void quant_kernel(const float* __restrict__ x, long long n,
                             const float* __restrict__ amax, int8_t* __restrict__ out) {
  float inv = 1.0f / qscale(*amax);
  long long stride = (long long)gridDim.x * blockDim.x;
  for (long long i = (long long)blockIdx.x * blockDim.x + threadIdx.x; i < n; i += stride) {
    int q = (int)rintf(x[i] * inv);
    q = q > 127 ? 127 : (q < -128 ? -128 : q);
    out[i] = (int8_t)q;
  }
}

__global__ void quant_v_kernel(const float* __restrict__ Vp,
                               const float* __restrict__ amax, int8_t* __restrict__ vt) {
  float inv = 1.0f / qscale(*amax);
  long long n = (long long)SQ * NKV * DH;
  long long stride = (long long)gridDim.x * blockDim.x;
  for (long long i = (long long)blockIdx.x * blockDim.x + threadIdx.x; i < n; i += stride) {
    int q = (int)rintf(Vp[i] * inv);
    q = q > 127 ? 127 : (q < -128 ? -128 : q);
    long long s = i / (NKV * DH);
    long long rem = i % (NKV * DH);
    vt[rem * SQ + s] = (int8_t)q;
  }
}

__global__ __launch_bounds__(32) void gemm_i8_kernel(
    const int8_t* __restrict__ A, const int8_t* __restrict__ B,
    float* __restrict__ C, int N, int K,
    const float* __restrict__ sA, const float* __restrict__ sB) {
  int lane = threadIdx.x & 31;
  int h = lane >> 4, l = lane & 15;
  int m0 = blockIdx.y << 4;
  int n0 = blockIdx.x << 6;
  v8i acc0 = v8i_zero(), acc1 = v8i_zero(), acc2 = v8i_zero(), acc3 = v8i_zero();
  const int8_t* arow = A + (size_t)(m0 + l) * K;
  const int8_t* b0p = B + (size_t)(n0 + 0 + l) * K;
  const int8_t* b1p = B + (size_t)(n0 + 16 + l) * K;
  const int8_t* b2p = B + (size_t)(n0 + 32 + l) * K;
  const int8_t* b3p = B + (size_t)(n0 + 48 + l) * K;

  v8i a  = load_a_frag_i8(arow, h);
  v8i b0 = load_b_frag_i8(b0p, h);
  v8i b1 = load_b_frag_i8(b1p, h);
  v8i b2 = load_b_frag_i8(b2p, h);
  v8i b3 = load_b_frag_i8(b3p, h);

  for (int k0 = 0; k0 < K - 64; k0 += 64) {
    int kn = k0 + 64;
    v8i an  = load_a_frag_i8(arow + kn, h);
    v8i nb0 = load_b_frag_i8(b0p + kn, h);
    v8i nb1 = load_b_frag_i8(b1p + kn, h);
    v8i nb2 = load_b_frag_i8(b2p + kn, h);
    v8i nb3 = load_b_frag_i8(b3p + kn, h);
    acc0 = __builtin_amdgcn_wmma_i32_16x16x64_iu8(true, a, true, b0, acc0, false, false);
    acc1 = __builtin_amdgcn_wmma_i32_16x16x64_iu8(true, a, true, b1, acc1, false, false);
    acc2 = __builtin_amdgcn_wmma_i32_16x16x64_iu8(true, a, true, b2, acc2, false, false);
    acc3 = __builtin_amdgcn_wmma_i32_16x16x64_iu8(true, a, true, b3, acc3, false, false);
    a = an; b0 = nb0; b1 = nb1; b2 = nb2; b3 = nb3;
  }
  acc0 = __builtin_amdgcn_wmma_i32_16x16x64_iu8(true, a, true, b0, acc0, false, false);
  acc1 = __builtin_amdgcn_wmma_i32_16x16x64_iu8(true, a, true, b1, acc1, false, false);
  acc2 = __builtin_amdgcn_wmma_i32_16x16x64_iu8(true, a, true, b2, acc2, false, false);
  acc3 = __builtin_amdgcn_wmma_i32_16x16x64_iu8(true, a, true, b3, acc3, false, false);

  float s = qscale(*sA) * qscale(*sB);
#pragma unroll
  for (int r = 0; r < 8; ++r) {
    size_t row = (size_t)(m0 + r + (h << 3)) * N;
    C[row + n0 + 0 + l]  = s * (float)acc0[r];
    C[row + n0 + 16 + l] = s * (float)acc1[r];
    C[row + n0 + 32 + l] = s * (float)acc2[r];
    C[row + n0 + 48 + l] = s * (float)acc3[r];
  }
}

__global__ __launch_bounds__(128) void rmsrope_kernel(
    const float* __restrict__ P, const float* __restrict__ w,
    const float* __restrict__ cs, const float* __restrict__ sn,
    float* __restrict__ out) {
  int s = blockIdx.x, hh = blockIdx.y, d = threadIdx.x;
  int nh = gridDim.y;
  __shared__ float xs[DH];
  __shared__ float red[4];
  float x = P[((size_t)s * nh + hh) * DH + d];
  xs[d] = x;
  float v = x * x;
#pragma unroll
  for (int off = 16; off; off >>= 1) v += __shfl_xor(v, off, 32);
  if ((d & 31) == 0) red[d >> 5] = v;
  __syncthreads();
  float var = (red[0] + red[1] + red[2] + red[3]) * (1.0f / DH);
  float inv = rsqrtf(var + 1e-5f);
  int pd = (d < 64) ? d + 64 : d - 64;
  float xn = x * inv * w[d];
  float xpn = xs[pd] * inv * w[pd];
  float rot = (d < 64) ? -xpn : xpn;
  out[((size_t)hh * SQ + s) * DH + d] = xn * cs[s * DH + d] + rot * sn[s * DH + d];
}

__global__ __launch_bounds__(256) void attn_kernel(
    const int8_t* __restrict__ q8, const int8_t* __restrict__ k8,
    const int8_t* __restrict__ v8t, const float* __restrict__ sc,
    float* __restrict__ out) {
  __shared__ __align__(16) int8_t pstage[8 * 16 * 64];
  int tid = threadIdx.x;
  int wave = tid >> 5, lane = tid & 31;
  int h = lane >> 4, l = lane & 15;
  int gw = blockIdx.x * 8 + wave;
  int hh = gw >> 7;
  int qt = gw & 127;
  int kvh = hh >> 2;
  float sscale = qscale(sc[5]) * qscale(sc[6]) * 0.08838834764831845f;
  float sv = qscale(sc[7]);

  const int8_t* qrow = q8 + ((size_t)hh * SQ + qt * 16 + l) * DH;
  v8i aq0 = load_a_frag_i8(qrow, h);
  v8i aq1 = load_a_frag_i8(qrow + 64, h);
  const int8_t* kbase = k8 + (size_t)kvh * SQ * DH;

  float Mx[8], Sm[8];
#pragma unroll
  for (int r = 0; r < 8; ++r) { Mx[r] = -3.0e38f; Sm[r] = 0.f; }

  {
    const int8_t* kcol0 = kbase + (size_t)l * DH;
    v8i b0 = load_b_frag_i8(kcol0, h);
    v8i b1 = load_b_frag_i8(kcol0 + 64, h);
    for (int kt = 0; kt <= qt; ++kt) {
      v8i nb0 = b0, nb1 = b1;
      if (kt < qt) {
        const int8_t* kcol = kbase + (size_t)((kt + 1) * 16 + l) * DH;
        nb0 = load_b_frag_i8(kcol, h);
        nb1 = load_b_frag_i8(kcol + 64, h);
      }
      v8i acc = v8i_zero();
      acc = __builtin_amdgcn_wmma_i32_16x16x64_iu8(true, aq0, true, b0, acc, false, false);
      acc = __builtin_amdgcn_wmma_i32_16x16x64_iu8(true, aq1, true, b1, acc, false, false);
      int key = kt * 16 + l;
#pragma unroll
      for (int r = 0; r < 8; ++r) {
        int m = r + (h << 3);
        float sv32 = (key <= qt * 16 + m) ? sscale * (float)acc[r] : -3.0e38f;
        float tm = sv32;
#pragma unroll
        for (int off = 8; off; off >>= 1) tm = fmaxf(tm, __shfl_xor(tm, off, 32));
        float nm = fmaxf(Mx[r], tm);
        float e = __expf(sv32 - nm);
        float ts = e;
#pragma unroll
        for (int off = 8; off; off >>= 1) ts += __shfl_xor(ts, off, 32);
        Sm[r] = Sm[r] * __expf(Mx[r] - nm) + ts;
        Mx[r] = nm;
      }
      b0 = nb0; b1 = nb1;
    }
  }

  v8i oacc[8];
#pragma unroll
  for (int t = 0; t < 8; ++t) oacc[t] = v8i_zero();
  float p127[8];
#pragma unroll
  for (int r = 0; r < 8; ++r) p127[r] = 127.0f / Sm[r];

  int8_t* lp = pstage + wave * 1024;
  int nchunk = (qt + 4) >> 2;
  for (int c = 0; c < nchunk; ++c) {
    int key0 = c * 64;
    v8i bv[8];
#pragma unroll
    for (int t = 0; t < 8; ++t) {
      const int8_t* vcol = v8t + ((size_t)kvh * DH + t * 16 + l) * SQ + key0;
      bv[t] = load_b_frag_i8(vcol, h);
    }
#pragma unroll
    for (int j = 0; j < 4; ++j) {
      int kt = c * 4 + j;
      if (kt <= qt) {
        const int8_t* kcol = kbase + (size_t)(kt * 16 + l) * DH;
        v8i b0 = load_b_frag_i8(kcol, h);
        v8i b1 = load_b_frag_i8(kcol + 64, h);
        v8i acc = v8i_zero();
        acc = __builtin_amdgcn_wmma_i32_16x16x64_iu8(true, aq0, true, b0, acc, false, false);
        acc = __builtin_amdgcn_wmma_i32_16x16x64_iu8(true, aq1, true, b1, acc, false, false);
        int key = kt * 16 + l;
#pragma unroll
        for (int r = 0; r < 8; ++r) {
          int m = r + (h << 3);
          float sv32 = (key <= qt * 16 + m) ? sscale * (float)acc[r] : -3.0e38f;
          int qv = (int)rintf(__expf(sv32 - Mx[r]) * p127[r]);
          qv = qv > 127 ? 127 : (qv < -128 ? -128 : qv);
          lp[m * 64 + j * 16 + l] = (int8_t)qv;
        }
      } else {
#pragma unroll
        for (int r = 0; r < 8; ++r) lp[(r + (h << 3)) * 64 + j * 16 + l] = 0;
      }
    }
    v8i ap = load_a_frag_i8(lp + l * 64, h);
#pragma unroll
    for (int t = 0; t < 8; ++t) {
      oacc[t] = __builtin_amdgcn_wmma_i32_16x16x64_iu8(true, ap, true, bv[t], oacc[t], false, false);
    }
  }
  float spv = (1.0f / 127.0f) * sv;
#pragma unroll
  for (int t = 0; t < 8; ++t) {
#pragma unroll
    for (int r = 0; r < 8; ++r) {
      int m = r + (h << 3);
      out[(size_t)(qt * 16 + m) * HIDN + hh * DH + t * 16 + l] = spv * (float)oacc[t][r];
    }
  }
}

extern "C" void kernel_launch(void* const* d_in, const int* in_sizes, int n_in,
                              void* d_out, int out_size, void* d_ws, size_t ws_size,
                              hipStream_t stream) {
  (void)in_sizes; (void)n_in; (void)out_size; (void)ws_size;
  const float* hidden = (const float*)d_in[0];
  const float* cosb   = (const float*)d_in[1];
  const float* sinb   = (const float*)d_in[2];
  const float* Wq = (const float*)d_in[4];
  const float* Wk = (const float*)d_in[5];
  const float* Wv = (const float*)d_in[6];
  const float* Wo = (const float*)d_in[7];
  const float* qnw = (const float*)d_in[8];
  const float* knw = (const float*)d_in[9];
  float* outp = (float*)d_out;

  char* ws = (char*)d_ws;
  unsigned* scU = (unsigned*)ws;
  float* scF = (float*)ws;
  size_t off = 256;
  auto alloc = [&](size_t bytes) {
    char* p = ws + off;
    off += (bytes + 255) & ~(size_t)255;
    return p;
  };
  int8_t* x8  = (int8_t*)alloc((size_t)SQ * HIDN);
  int8_t* Wq8 = (int8_t*)alloc((size_t)HIDN * HIDN);
  int8_t* Wk8 = (int8_t*)alloc((size_t)NKV * DH * HIDN);
  int8_t* Wv8 = (int8_t*)alloc((size_t)NKV * DH * HIDN);
  int8_t* Wo8 = (int8_t*)alloc((size_t)HIDN * HIDN);
  float*  Qp  = (float*)alloc((size_t)SQ * HIDN * 4);
  float*  Kp  = (float*)alloc((size_t)SQ * NKV * DH * 4);
  float*  Vp  = (float*)alloc((size_t)SQ * NKV * DH * 4);
  float*  qr  = (float*)alloc((size_t)NH * SQ * DH * 4);
  float*  kr  = (float*)alloc((size_t)NKV * SQ * DH * 4);
  int8_t* q8  = (int8_t*)alloc((size_t)NH * SQ * DH);
  int8_t* k8  = (int8_t*)alloc((size_t)NKV * SQ * DH);
  int8_t* v8t = (int8_t*)alloc((size_t)NKV * DH * SQ);
  int8_t* a8  = (int8_t*)alloc((size_t)SQ * HIDN);
  float*  aout = Qp;

  init_scalars_kernel<<<1, 32, 0, stream>>>(scU);
  absmax_kernel<<<512, 256, 0, stream>>>(hidden, (long long)SQ * HIDN, scU + 0);
  absmax_kernel<<<512, 256, 0, stream>>>(Wq, (long long)HIDN * HIDN, scU + 1);
  absmax_kernel<<<512, 256, 0, stream>>>(Wk, (long long)NKV * DH * HIDN, scU + 2);
  absmax_kernel<<<512, 256, 0, stream>>>(Wv, (long long)NKV * DH * HIDN, scU + 3);
  absmax_kernel<<<512, 256, 0, stream>>>(Wo, (long long)HIDN * HIDN, scU + 4);

  quant_kernel<<<1024, 256, 0, stream>>>(hidden, (long long)SQ * HIDN, scF + 0, x8);
  quant_kernel<<<1024, 256, 0, stream>>>(Wq, (long long)HIDN * HIDN, scF + 1, Wq8);
  quant_kernel<<<1024, 256, 0, stream>>>(Wk, (long long)NKV * DH * HIDN, scF + 2, Wk8);
  quant_kernel<<<1024, 256, 0, stream>>>(Wv, (long long)NKV * DH * HIDN, scF + 3, Wv8);
  quant_kernel<<<1024, 256, 0, stream>>>(Wo, (long long)HIDN * HIDN, scF + 4, Wo8);

  gemm_i8_kernel<<<dim3(HIDN / 64, SQ / 16), 32, 0, stream>>>(x8, Wq8, Qp, HIDN, HIDN, scF + 0, scF + 1);
  gemm_i8_kernel<<<dim3((NKV * DH) / 64, SQ / 16), 32, 0, stream>>>(x8, Wk8, Kp, NKV * DH, HIDN, scF + 0, scF + 2);
  gemm_i8_kernel<<<dim3((NKV * DH) / 64, SQ / 16), 32, 0, stream>>>(x8, Wv8, Vp, NKV * DH, HIDN, scF + 0, scF + 3);

  rmsrope_kernel<<<dim3(SQ, NH), DH, 0, stream>>>(Qp, qnw, cosb, sinb, qr);
  rmsrope_kernel<<<dim3(SQ, NKV), DH, 0, stream>>>(Kp, knw, cosb, sinb, kr);

  absmax_kernel<<<512, 256, 0, stream>>>(qr, (long long)NH * SQ * DH, scU + 5);
  absmax_kernel<<<512, 256, 0, stream>>>(kr, (long long)NKV * SQ * DH, scU + 6);
  absmax_kernel<<<512, 256, 0, stream>>>(Vp, (long long)SQ * NKV * DH, scU + 7);

  quant_kernel<<<1024, 256, 0, stream>>>(qr, (long long)NH * SQ * DH, scF + 5, q8);
  quant_kernel<<<1024, 256, 0, stream>>>(kr, (long long)NKV * SQ * DH, scF + 6, k8);
  quant_v_kernel<<<1024, 256, 0, stream>>>(Vp, scF + 7, v8t);

  attn_kernel<<<(NH * (SQ / 16)) / 8, 256, 0, stream>>>(q8, k8, v8t, scF, aout);

  absmax_kernel<<<512, 256, 0, stream>>>(aout, (long long)SQ * HIDN, scU + 8);
  quant_kernel<<<1024, 256, 0, stream>>>(aout, (long long)SQ * HIDN, scF + 8, a8);

  gemm_i8_kernel<<<dim3(HIDN / 64, SQ / 16), 32, 0, stream>>>(a8, Wo8, outp, HIDN, HIDN, scF + 8, scF + 4);
}
